// SpliceMambaV5_35158602285288
// MI455X (gfx1250) — matrix operand order repaired and checked by reading, not compiled, not run on hardware
//
#include <hip/hip_runtime.h>
#include <cstddef>

// ---------------- problem constants ----------------
#define BB 2
#define LL 4096
#define DD 256
#define HH 8
#define NCC 3
#define TOPN 8
#define RR 32
#define LSS 1000
#define LEE 3000
#define NLL 2
#define FFD 1024
#define QMAX 1040          // 2*TOPN*(2R+1)
#define HD 32
#define QROWS (BB*QMAX)    // 2080
#define KVROWS (BB*LL)     // 8192
#define REGION (LEE-LSS)   // 2000

typedef __attribute__((ext_vector_type(16))) __bf16 v16bf;
typedef __attribute__((ext_vector_type(8)))  float  v8f;
typedef __attribute__((ext_vector_type(4)))  float  f4;

static __device__ inline v8f wmma_bf16(v16bf a, v16bf b, v8f c) {
  return __builtin_amdgcn_wmma_f32_16x16x32_bf16(false, a, false, b, (short)0, c,
                                                 false, false);
}

// ---- CDNA5 async memory->LDS mover (ASYNCcnt tracked, no VGPR round-trip) --
// LDS destination address is taken from a VGPR (plus the wave's LDS base); generic shared pointers carry the LDS
// offset in their low 32 bits.
static __device__ inline void async_ld_b128(void* lds_dst, const void* gsrc) {
  unsigned loff = (unsigned)(size_t)lds_dst;
  unsigned long long ga = (unsigned long long)(size_t)gsrc;
  asm volatile("global_load_async_to_lds_b128 %0, %1, off"
               :: "v"(loff), "v"(ga) : "memory");
}
static __device__ inline void wait_async() {
  asm volatile("s_wait_asynccnt 0x0" ::: "memory");
}

// convert four float4 runs into a bf16 fragment (elements in load order)
static __device__ inline v16bf cvt_frag(f4 p0, f4 p1, f4 p2, f4 p3) {
  v16bf r;
  r[0]  = (__bf16)p0[0]; r[1]  = (__bf16)p0[1]; r[2]  = (__bf16)p0[2]; r[3]  = (__bf16)p0[3];
  r[4]  = (__bf16)p1[0]; r[5]  = (__bf16)p1[1]; r[6]  = (__bf16)p1[2]; r[7]  = (__bf16)p1[3];
  r[8]  = (__bf16)p2[0]; r[9]  = (__bf16)p2[1]; r[10] = (__bf16)p2[2]; r[11] = (__bf16)p2[3];
  r[12] = (__bf16)p3[0]; r[13] = (__bf16)p3[1]; r[14] = (__bf16)p3[2]; r[15] = (__bf16)p3[3];
  return r;
}

// A fragment (16x32 bf16): element j -> k = 8*hi + (j&7) + ((j&8)?16:0);
// base must already include + 8*hi.
static __device__ inline v16bf load_a_frag_fast(const float* base) {
  f4 p0 = *(const f4*)(base);
  f4 p1 = *(const f4*)(base + 4);
  f4 p2 = *(const f4*)(base + 16);
  f4 p3 = *(const f4*)(base + 20);
  return cvt_frag(p0, p1, p2, p3);
}

// ---------------- device scratch pool (avoids ws_size dependence) -----------
#define O_KVLN  ((size_t)0)                    // 8192*256
#define O_QLN   (O_KVLN + (size_t)KVROWS*DD)
#define O_Q     (O_QLN  + (size_t)QROWS*DD)
#define O_K     (O_Q    + (size_t)QROWS*DD)
#define O_V     (O_K    + (size_t)KVROWS*DD)
#define O_QS    (O_V    + (size_t)KVROWS*DD)
#define O_AO    (O_QS   + (size_t)QROWS*DD)
#define O_H1    (O_AO   + (size_t)QROWS*DD)    // 2080*1024
#define O_REF   (O_H1   + (size_t)QROWS*FFD)
#define O_IDX   (O_REF  + (size_t)KVROWS*DD)   // 2080 ints
#define O_MASK  (O_IDX  + (size_t)QROWS)
#define POOL_F  (O_MASK + (size_t)QROWS)

__device__ float g_pool[POOL_F];

// ---------------- top-k vicinity selection + sort/dedupe --------------------
__global__ void select_kernel(const float* __restrict__ logits,
                              int* __restrict__ g_idx, int* __restrict__ g_mask) {
  const int b = blockIdx.x, tid = threadIdx.x;
  __shared__ float don[REGION];
  __shared__ float acc[REGION];
  __shared__ int   centers[16];
  __shared__ float redv[256];
  __shared__ int   redi[256];
  __shared__ int   s[2048];

  for (int p = tid; p < REGION; p += 256) {
    const float* x = logits + ((size_t)b * LL + (LSS + p)) * NCC;
    float m = fmaxf(x[0], fmaxf(x[1], x[2]));
    float lse = m + logf(expf(x[0] - m) + expf(x[1] - m) + expf(x[2] - m));
    acc[p] = x[1] - lse;   // acceptor log-prob
    don[p] = x[2] - lse;   // donor log-prob
  }
  __syncthreads();

  for (int sel = 0; sel < 16; ++sel) {
    float* arr = (sel < 8) ? don : acc;   // donors first (reference concat order)
    float bv = -3.0e38f; int bi = 0x7fffffff;
    for (int p = tid; p < REGION; p += 256) {
      float v = arr[p];
      if (v > bv || (v == bv && p < bi)) { bv = v; bi = p; }
    }
    redv[tid] = bv; redi[tid] = bi;
    __syncthreads();
    for (int st = 128; st > 0; st >>= 1) {
      if (tid < st) {
        if (redv[tid + st] > redv[tid] ||
            (redv[tid + st] == redv[tid] && redi[tid + st] < redi[tid])) {
          redv[tid] = redv[tid + st]; redi[tid] = redi[tid + st];
        }
      }
      __syncthreads();
    }
    if (tid == 0) { centers[sel] = LSS + redi[0]; arr[redi[0]] = -3.0e38f; }
    __syncthreads();
  }

  for (int j = tid; j < 2048; j += 256) {
    if (j < QMAX) {
      int c = j / (2 * RR + 1), o = j % (2 * RR + 1) - RR;
      int v = centers[c] + o;
      v = v < 0 ? 0 : (v > LL - 1 ? LL - 1 : v);
      s[j] = v;
    } else {
      s[j] = 0x7fffffff;
    }
  }
  __syncthreads();

  for (int k = 2; k <= 2048; k <<= 1) {
    for (int jj = k >> 1; jj > 0; jj >>= 1) {
      for (int i = tid; i < 2048; i += 256) {
        int ixj = i ^ jj;
        if (ixj > i) {
          bool up = (i & k) == 0;
          int a = s[i], c = s[ixj];
          if ((a > c) == up) { s[i] = c; s[ixj] = a; }
        }
      }
      __syncthreads();
    }
  }

  for (int j = tid; j < QMAX; j += 256) {
    int v = s[j];
    int first = (j == 0) || (s[j - 1] != v);
    g_idx[b * QMAX + j]  = first ? v : LL;
    g_mask[b * QMAX + j] = first;
  }
}

// ---------------- gather / copy / scatter -----------------------------------
__global__ void gather_kernel(const float* __restrict__ enc,
                              const int* __restrict__ idx, float* __restrict__ qs) {
  int r = blockIdx.x, d = threadIdx.x;
  int b = r / QMAX;
  int ix = idx[r];
  qs[(size_t)r * DD + d] = (ix < LL) ? enc[((size_t)b * LL + ix) * DD + d] : 0.f;
}

__global__ void copy_kernel(const float* __restrict__ src, float* __restrict__ dst) {
  size_t o = ((size_t)blockIdx.x * 256 + threadIdx.x) * 4;
  *(f4*)(dst + o) = *(const f4*)(src + o);
}

__global__ void scatter_kernel(const float* __restrict__ qs,
                               const int* __restrict__ idx, float* __restrict__ ref) {
  int r = blockIdx.x, d = threadIdx.x;
  int b = r / QMAX;
  int ix = idx[r];
  if (ix < LL) ref[((size_t)b * LL + ix) * DD + d] = qs[(size_t)r * DD + d];
}

// ---------------- layer norm (D = 256, block = 256) -------------------------
__global__ void ln_kernel(const float* __restrict__ x, const float* __restrict__ g,
                          const float* __restrict__ bta, float* __restrict__ y) {
  int r = blockIdx.x, tid = threadIdx.x;
  __shared__ float red[256];
  float v = x[(size_t)r * DD + tid];
  red[tid] = v; __syncthreads();
  for (int st = 128; st > 0; st >>= 1) { if (tid < st) red[tid] += red[tid + st]; __syncthreads(); }
  float mean = red[0] * (1.f / DD);
  __syncthreads();
  float d = v - mean;
  red[tid] = d * d; __syncthreads();
  for (int st = 128; st > 0; st >>= 1) { if (tid < st) red[tid] += red[tid + st]; __syncthreads(); }
  float var = red[0] * (1.f / DD);
  y[(size_t)r * DD + tid] = d * rsqrtf(var + 1e-5f) * g[tid] + bta[tid];
}

// ---------------- WMMA GEMM: C = epi(A(MxK) @ W(KxN) + bias) ----------------
// Requires M, N, K all multiples of 32 (true for every call site here).
#define EPI_NONE 0
#define EPI_GELU 1
#define EPI_RESID_MASK 2

template <int EPI>
__launch_bounds__(32)
__global__ void gemm_kernel(const float* __restrict__ A, const float* __restrict__ W,
                            const float* __restrict__ bias, float* __restrict__ C,
                            const float* __restrict__ resid, const int* __restrict__ mask,
                            int N, int K) {
  const int lane = threadIdx.x;
  const int lo = lane & 15, hi = lane >> 4;
  const int n0 = blockIdx.x * 32;
  const int m0 = blockIdx.y * 32;
  __shared__ float lb[32][32];   // W tile [k][n]; 16B-aligned rows for async b128

  v8f a00 = {}, a01 = {}, a10 = {}, a11 = {};
  for (int k0 = 0; k0 < K; k0 += 32) {
    // stage W tile via async mover: one coalesced row (8x async b128) per lane
    {
      const float* wr = W + (size_t)(k0 + lane) * N + n0;
#pragma unroll
      for (int c = 0; c < 32; c += 4) async_ld_b128(&lb[lane][c], wr + c);
      if (k0 + 32 < K) __builtin_prefetch(wr + (size_t)32 * N, 0, 0);
    }
    // A fragments straight from global (b128 loads) while async copies fly
    v16bf fa0 = load_a_frag_fast(A + (size_t)(m0 + lo) * K + k0 + 8 * hi);
    v16bf fa1 = load_a_frag_fast(A + (size_t)(m0 + 16 + lo) * K + k0 + 8 * hi);
    wait_async();
    __syncthreads();
    // B fragments from LDS columns: element j -> k = 8*hi + (j&7) + ((j&8)?16:0)
    v16bf fb0, fb1;
#pragma unroll
    for (int j = 0; j < 16; ++j) {
      int kk = (j & 7) + 8 * hi + ((j & 8) ? 16 : 0);
      fb0[j] = (__bf16)lb[kk][lo];
      fb1[j] = (__bf16)lb[kk][16 + lo];
    }
    a00 = wmma_bf16(fa0, fb0, a00);
    a01 = wmma_bf16(fa0, fb1, a01);
    a10 = wmma_bf16(fa1, fb0, a10);
    a11 = wmma_bf16(fa1, fb1, a11);
    __syncthreads();   // protect lb before next staging
  }

  v8f accs[2][2] = {a00, a01, a10, a11};
#pragma unroll
  for (int tm = 0; tm < 2; ++tm)
#pragma unroll
    for (int tn = 0; tn < 2; ++tn)
#pragma unroll
      for (int i = 0; i < 8; ++i) {
        int row = m0 + tm * 16 + i + 8 * hi;
        int col = n0 + tn * 16 + lo;
        float v = accs[tm][tn][i] + bias[col];
        if (EPI == EPI_GELU) {
          v = 0.5f * v * (1.f + erff(v * 0.70710678118654752f));
        } else if (EPI == EPI_RESID_MASK) {
          v = mask[row] ? (resid[(size_t)row * N + col] + v) : 0.f;
        }
        C[(size_t)row * N + col] = v;
      }
}

// ---------------- tiny head GEMM (N = 3), plain VALU ------------------------
__global__ void head_kernel(const float* __restrict__ ref, const float* __restrict__ hw,
                            const float* __restrict__ hb, float* __restrict__ out) {
  int r = blockIdx.x * blockDim.x + threadIdx.x;   // 8192 rows
  const float* row = ref + (size_t)r * DD;
  float a0 = hb[0], a1 = hb[1], a2 = hb[2];
  for (int d = 0; d < DD; d += 4) {
    f4 x = *(const f4*)(row + d);
#pragma unroll
    for (int c = 0; c < 4; ++c) {
      float xv = x[c];
      a0 += xv * hw[(d + c) * NCC + 0];
      a1 += xv * hw[(d + c) * NCC + 1];
      a2 += xv * hw[(d + c) * NCC + 2];
    }
  }
  out[(size_t)r * NCC + 0] = a0;
  out[(size_t)r * NCC + 1] = a1;
  out[(size_t)r * NCC + 2] = a2;
}

// ---------------- flash attention, one wave per 16-row Q tile ---------------
__launch_bounds__(32)
__global__ void attn_kernel(const float* __restrict__ q, const float* __restrict__ k,
                            const float* __restrict__ v, float* __restrict__ ao) {
  const int lane = threadIdx.x;
  const int qt = blockIdx.x, h = blockIdx.y, b = blockIdx.z;
  const int hi = lane >> 4, lo = lane & 15;
  __shared__ float  sc[16][32];
  __shared__ __bf16 pl[16][32];
  __shared__ float  lv[32][32];   // 16B-aligned rows for async b128
  __shared__ float  alpha_s[16];
  __shared__ float  linv_s[16];

  const float scale = 0.17677669529663687f;  // 1/sqrt(HD)

  // Q fragment (A layout, K = HD = 32), pre-scaled, b128 loads
  v16bf aq;
  {
    const float* base = q + ((size_t)b * QMAX + qt * 16 + lo) * DD + h * HD + 16 * hi;
    f4 p0 = *(const f4*)(base);
    f4 p1 = *(const f4*)(base + 4);
    f4 p2 = *(const f4*)(base + 8);
    f4 p3 = *(const f4*)(base + 12);
#pragma unroll
    for (int c = 0; c < 4; ++c) {
      aq[c]      = (__bf16)(p0[c] * scale);
      aq[4 + c]  = (__bf16)(p1[c] * scale);
      aq[8 + c]  = (__bf16)(p2[c] * scale);
      aq[12 + c] = (__bf16)(p3[c] * scale);
    }
  }

  v8f o0 = {}, o1 = {};
  float m_r = -1.0e30f, l_r = 0.f;

  for (int kc = 0; kc < LL; kc += 32) {
    // stage V tile [key][dim] via async mover: one coalesced row per lane
    {
      const float* vr = v + ((size_t)b * LL + kc + lane) * DD + h * HD;
#pragma unroll
      for (int c = 0; c < 32; c += 4) async_ld_b128(&lv[lane][c], vr + c);
    }
    // K^T fragments (B layout): element j -> d = j + 16*hi, contiguous b128
    v16bf bk0, bk1;
    {
      const float* k0p = k + ((size_t)b * LL + kc + lo) * DD + h * HD + 16 * hi;
      const float* k1p = k + ((size_t)b * LL + kc + 16 + lo) * DD + h * HD + 16 * hi;
      bk0 = cvt_frag(*(const f4*)(k0p), *(const f4*)(k0p + 4),
                     *(const f4*)(k0p + 8), *(const f4*)(k0p + 12));
      bk1 = cvt_frag(*(const f4*)(k1p), *(const f4*)(k1p + 4),
                     *(const f4*)(k1p + 8), *(const f4*)(k1p + 12));
    }
    v8f c0 = {}, c1 = {};
    c0 = wmma_bf16(aq, bk0, c0);
    c1 = wmma_bf16(aq, bk1, c1);

#pragma unroll
    for (int i = 0; i < 8; ++i) {
      int r = i + 8 * hi;
      sc[r][lo]      = c0[i];
      sc[r][16 + lo] = c1[i];
    }
    wait_async();
    __syncthreads();   // sc and lv visible

    if (lane < 16) {
      float cmax = -1.0e30f;
      for (int c = 0; c < 32; ++c) cmax = fmaxf(cmax, sc[lane][c]);
      float mnew = fmaxf(m_r, cmax);
      float al = expf(m_r - mnew);
      float ssum = 0.f;
      for (int c = 0; c < 32; ++c) {
        float p = expf(sc[lane][c] - mnew);
        ssum += p;
        pl[lane][c] = (__bf16)p;
      }
      l_r = l_r * al + ssum;
      m_r = mnew;
      alpha_s[lane] = al;
    }
    __syncthreads();

#pragma unroll
    for (int i = 0; i < 8; ++i) {
      float al = alpha_s[i + 8 * hi];
      o0[i] *= al; o1[i] *= al;
    }

    // P (A layout) from LDS, V (B layout) from LDS tile
    v16bf ap, bv0, bv1;
#pragma unroll
    for (int j = 0; j < 16; ++j) {
      int kk = j + 16 * hi;
      ap[j] = pl[lo][kk];
      int kv = j + 16 * hi;
      bv0[j] = (__bf16)lv[kv][lo];
      bv1[j] = (__bf16)lv[kv][16 + lo];
    }
    o0 = wmma_bf16(ap, bv0, o0);
    o1 = wmma_bf16(ap, bv1, o1);
    __syncthreads();   // protect sc/pl/lv before next chunk
  }

  if (lane < 16) linv_s[lane] = 1.f / l_r;
  __syncthreads();

#pragma unroll
  for (int i = 0; i < 8; ++i) {
    float inv = linv_s[i + 8 * hi];
    size_t row = ((size_t)b * QMAX + qt * 16 + i + 8 * hi) * DD + h * HD;
    ao[row + lo]      = o0[i] * inv;
    ao[row + 16 + lo] = o1[i] * inv;
  }
}

// ---------------- host-side orchestration -----------------------------------
extern "C" void kernel_launch(void* const* d_in, const int* in_sizes, int n_in,
                              void* d_out, int out_size, void* d_ws, size_t ws_size,
                              hipStream_t stream) {
  (void)in_sizes; (void)n_in; (void)out_size; (void)d_ws; (void)ws_size;
  const float* enc     = (const float*)d_in[0];
  const float* logits  = (const float*)d_in[1];
  const float* ln_q_g  = (const float*)d_in[2];
  const float* ln_q_b  = (const float*)d_in[3];
  const float* ln_kv_g = (const float*)d_in[4];
  const float* ln_kv_b = (const float*)d_in[5];
  const float* Wq = (const float*)d_in[6];
  const float* bq = (const float*)d_in[7];
  const float* Wk = (const float*)d_in[8];
  const float* bk = (const float*)d_in[9];
  const float* Wv = (const float*)d_in[10];
  const float* bv = (const float*)d_in[11];
  const float* Wo = (const float*)d_in[12];
  const float* bo = (const float*)d_in[13];
  const float* ffn_g = (const float*)d_in[14];
  const float* ffn_b = (const float*)d_in[15];
  const float* W1 = (const float*)d_in[16];
  const float* b1 = (const float*)d_in[17];
  const float* W2 = (const float*)d_in[18];
  const float* b2 = (const float*)d_in[19];
  const float* headW = (const float*)d_in[20];
  const float* headb = (const float*)d_in[21];
  float* out = (float*)d_out;

  void* poolv = nullptr;
  hipGetSymbolAddress(&poolv, HIP_SYMBOL(g_pool));
  float* pool = (float*)poolv;
  float* kvln = pool + O_KVLN;
  float* qln  = pool + O_QLN;
  float* qbuf = pool + O_Q;
  float* kbuf = pool + O_K;
  float* vbuf = pool + O_V;
  float* qs   = pool + O_QS;
  float* aob  = pool + O_AO;
  float* h1   = pool + O_H1;
  float* ref  = pool + O_REF;
  int*   idxp = (int*)(pool + O_IDX);
  int*   mskp = (int*)(pool + O_MASK);

  select_kernel<<<BB, 256, 0, stream>>>(logits, idxp, mskp);
  gather_kernel<<<QROWS, 256, 0, stream>>>(enc, idxp, qs);

  for (int i = 0; i < NLL; ++i) {
    const size_t wo = (size_t)i * DD * DD;
    const size_t fo = (size_t)i * DD * FFD;
    ln_kernel<<<KVROWS, 256, 0, stream>>>(enc, ln_kv_g + i * DD, ln_kv_b + i * DD, kvln);
    ln_kernel<<<QROWS, 256, 0, stream>>>(qs, ln_q_g + i * DD, ln_q_b + i * DD, qln);
    gemm_kernel<EPI_NONE><<<dim3(DD / 32, QROWS / 32), 32, 0, stream>>>(
        qln, Wq + wo, bq + i * DD, qbuf, nullptr, nullptr, DD, DD);
    gemm_kernel<EPI_NONE><<<dim3(DD / 32, KVROWS / 32), 32, 0, stream>>>(
        kvln, Wk + wo, bk + i * DD, kbuf, nullptr, nullptr, DD, DD);
    gemm_kernel<EPI_NONE><<<dim3(DD / 32, KVROWS / 32), 32, 0, stream>>>(
        kvln, Wv + wo, bv + i * DD, vbuf, nullptr, nullptr, DD, DD);
    attn_kernel<<<dim3(QMAX / 16, HH, BB), 32, 0, stream>>>(qbuf, kbuf, vbuf, aob);
    gemm_kernel<EPI_RESID_MASK><<<dim3(DD / 32, QROWS / 32), 32, 0, stream>>>(
        aob, Wo + wo, bo + i * DD, qs, qs, mskp, DD, DD);
    ln_kernel<<<QROWS, 256, 0, stream>>>(qs, ffn_g + i * DD, ffn_b + i * DD, qln);
    gemm_kernel<EPI_GELU><<<dim3(FFD / 32, QROWS / 32), 32, 0, stream>>>(
        qln, W1 + fo, b1 + i * FFD, h1, nullptr, nullptr, FFD, DD);
    gemm_kernel<EPI_RESID_MASK><<<dim3(DD / 32, QROWS / 32), 32, 0, stream>>>(
        h1, W2 + fo, b2 + i * DD, qs, qs, mskp, DD, FFD);
  }

  copy_kernel<<<KVROWS, 256, 0, stream>>>(enc, ref);
  scatter_kernel<<<QROWS, 256, 0, stream>>>(qs, idxp, ref);
  head_kernel<<<KVROWS / 256, 256, 0, stream>>>(ref, headW, headb, out);
}
